// AttentionWithEinops_6545530159234
// MI455X (gfx1250) — matrix operand order repaired and checked by reading, not compiled, not run on hardware
//
#include <hip/hip_runtime.h>

using bf16 = __bf16;
typedef bf16  v16bf __attribute__((ext_vector_type(16)));
typedef bf16  v8bf  __attribute__((ext_vector_type(8)));
typedef bf16  v4bf  __attribute__((ext_vector_type(4)));
typedef float v8f   __attribute__((ext_vector_type(8)));

#define NB 4
#define NS 2048
#define ND 2048
#define NH 16
#define NE 128
#define NHE (NH * NE)   // 2048
#define LSTR 40         // LDS row stride in bf16 elems (32 + 8 pad), keeps 16B alignment

__device__ __forceinline__ v16bf cat16(v8bf lo, v8bf hi) {
  v16bf r;
#pragma unroll
  for (int i = 0; i < 8; ++i) { r[i] = lo[i]; r[i + 8] = hi[i]; }
  return r;
}

__device__ __forceinline__ v8f wmma_bf16(v16bf a, v16bf b, v8f c) {
  // D = A(16x32 bf16) * B(32x16 bf16) + C(16x16 f32)
  return __builtin_amdgcn_wmma_f32_16x16x32_bf16(false, a, false, b, (short)0, c,
                                                 false, false);
}

// CDNA5 async copy: 16B per lane, global -> LDS, tracked by ASYNCcnt.
// lds_off = wave-relative LDS byte address (low 32 bits of a generic LDS pointer).
__device__ __forceinline__ void async_ld_b128(unsigned lds_off, const void* gaddr) {
  asm volatile("global_load_async_to_lds_b128 %0, %1, off"
               :: "v"(lds_off), "v"(gaddr)
               : "memory");
}
__device__ __forceinline__ void wait_async0() {
  asm volatile("s_wait_asynccnt 0" ::: "memory");
}

// ---------------------------------------------------------------------------
// f32 -> bf16 elementwise convert (vector x4)
// ---------------------------------------------------------------------------
__global__ void cvt_bf16_kernel(const float* __restrict__ in, bf16* __restrict__ out,
                                int n4) {
  int i = blockIdx.x * blockDim.x + threadIdx.x;
  if (i < n4) {
    float4 f = ((const float4*)in)[i];
    v4bf o;
    o[0] = (bf16)f.x; o[1] = (bf16)f.y; o[2] = (bf16)f.z; o[3] = (bf16)f.w;
    ((v4bf*)out)[i] = o;
  }
}

// ---------------------------------------------------------------------------
// W_O [he][d] f32 -> WoT [d][he] bf16 (tiled transpose through LDS)
// ---------------------------------------------------------------------------
__global__ void cvt_t_kernel(const float* __restrict__ in, bf16* __restrict__ out) {
  __shared__ float tile[32][33];
  const int x = blockIdx.x * 32 + threadIdx.x;  // d
  const int y0 = blockIdx.y * 32;               // he
#pragma unroll
  for (int i = 0; i < 32; i += 8)
    tile[threadIdx.y + i][threadIdx.x] = in[(size_t)(y0 + threadIdx.y + i) * ND + x];
  __syncthreads();
  const int xo = blockIdx.y * 32 + threadIdx.x; // he
  const int yo = blockIdx.x * 32;               // d
#pragma unroll
  for (int i = 0; i < 32; i += 8)
    out[(size_t)(yo + threadIdx.y + i) * NHE + xo] = (bf16)tile[threadIdx.x][threadIdx.y + i];
}

// ---------------------------------------------------------------------------
// One 128x128x32 tile step: 8 WMMAs per wave from LDS fragments.
// A-operand lane layout used here: lane<16 -> K {0..15}, lane>=16 -> K {16..31} (M = lane&15).
// B-operand lane layout: lane<16 -> K {0..15}, lane>=16 -> K {16..31} (N = lane&15).
// ---------------------------------------------------------------------------
__device__ __forceinline__ void gemm_tile_compute(const bf16* curA, const bf16* curB,
                                                  int wm, int wn, int lrow, int lhalf,
                                                  v8f acc[4][2]) {
  v16bf af[4];
#pragma unroll
  for (int mf = 0; mf < 4; ++mf) {
    const bf16* pA = curA + (wm * 64 + mf * 16 + lrow) * LSTR + lhalf * 16;
    af[mf] = cat16(*(const v8bf*)pA, *(const v8bf*)(pA + 8));
  }
  v16bf bfr[2];
#pragma unroll
  for (int nf = 0; nf < 2; ++nf) {
    const bf16* pB = curB + (wn * 32 + nf * 16 + lrow) * LSTR + lhalf * 16;
    bfr[nf] = cat16(*(const v8bf*)pB, *(const v8bf*)(pB + 8));
  }
#pragma unroll
  for (int mf = 0; mf < 4; ++mf)
#pragma unroll
    for (int nf = 0; nf < 2; ++nf)
      acc[mf][nf] = wmma_bf16(af[mf], bfr[nf], acc[mf][nf]);
}

// ---------------------------------------------------------------------------
// Tiled GEMM core: C[128x128] += A[128xK] * B^T[128xK], K = 2048, row-major,
// 256 threads = 8 waves (2 x 4), wave tile 64x32 = 4x2 wmma accumulators.
// Double-buffered LDS, tiles staged with GLOBAL_LOAD_ASYNC_TO_LDS_B128 so the
// copy of tile k+1 overlaps the WMMAs on tile k; one barrier per K-step.
// Last K-step is peeled so the steady-state loop has no inner branch.
// ---------------------------------------------------------------------------
__device__ __forceinline__ void gemm_core_2048(const bf16* __restrict__ A,
                                               const bf16* __restrict__ Bm,
                                               int mbase, int nbase, v8f acc[4][2]) {
  __shared__ bf16 sA[2][128 * LSTR];
  __shared__ bf16 sB[2][128 * LSTR];
  const int tid = threadIdx.x;
  const int lane = tid & 31, wid = tid >> 5;
  const int lrow = lane & 15, lhalf = lane >> 4;
  const int wm = wid & 1, wn = wid >> 1;
  const int ldr = tid >> 1, ldc = (tid & 1) * 16;
  const bf16* ga = A + (size_t)(mbase + ldr) * ND + ldc;
  const bf16* gb = Bm + (size_t)(nbase + ldr) * ND + ldc;
  const unsigned bufsz = 128 * LSTR * 2u;  // bytes per buffer
  const unsigned soff = (unsigned)(ldr * LSTR + ldc) * 2u;
  const unsigned saBase = (unsigned)(uintptr_t)&sA[0][0];  // LDS byte address
  const unsigned sbBase = (unsigned)(uintptr_t)&sB[0][0];

  // prologue: prefetch tile 0 into buffer 0
  async_ld_b128(saBase + soff, ga);
  async_ld_b128(saBase + soff + 16, ga + 8);
  async_ld_b128(sbBase + soff, gb);
  async_ld_b128(sbBase + soff + 16, gb + 8);
  ga += 32; gb += 32;

  int p = 0;
  for (int k0 = 0; k0 < ND - 32; k0 += 32) {
    wait_async0();       // this wave's async writes to buffer p are in LDS
    __syncthreads();     // => every wave's writes to buffer p are visible
    // prefetch next tile into buffer p^1 (overlaps the WMMAs below)
    const unsigned q = (unsigned)(p ^ 1) * bufsz;
    async_ld_b128(saBase + q + soff, ga);
    async_ld_b128(saBase + q + soff + 16, ga + 8);
    async_ld_b128(sbBase + q + soff, gb);
    async_ld_b128(sbBase + q + soff + 16, gb + 8);
    ga += 32; gb += 32;
    gemm_tile_compute(&sA[p][0], &sB[p][0], wm, wn, lrow, lhalf, acc);
    p ^= 1;
  }
  // peeled final K-step (no prefetch)
  wait_async0();
  __syncthreads();
  gemm_tile_compute(&sA[p][0], &sB[p][0], wm, wn, lrow, lhalf, acc);
}

// ---------------------------------------------------------------------------
// Fused Q/K/V projection. blockIdx.z selects which weight / epilogue.
// q: bf16 * (1/sqrt(128)); k: bf16 + f32 cache; v: f32 cache + bf16 V^T [b,h,e,s].
// ---------------------------------------------------------------------------
__global__ __launch_bounds__(256) void qkv_gemm_kernel(
    const bf16* __restrict__ A, const bf16* __restrict__ Wq,
    const bf16* __restrict__ Wk, const bf16* __restrict__ Wv,
    bf16* __restrict__ qb, bf16* __restrict__ kb, bf16* __restrict__ vT,
    float* __restrict__ kout, float* __restrict__ vout) {
  const int sel = blockIdx.z;
  const bf16* Bm = (sel == 0) ? Wq : (sel == 1 ? Wk : Wv);
  const int mbase = blockIdx.x * 128, nbase = blockIdx.y * 128;
  v8f acc[4][2] = {};
  gemm_core_2048(A, Bm, mbase, nbase, acc);
  const int lane = threadIdx.x & 31, wid = threadIdx.x >> 5;
  const int lrow = lane & 15, lhalf = lane >> 4;
  const int wm = wid & 1, wn = wid >> 1;
#pragma unroll
  for (int mf = 0; mf < 4; ++mf) {
#pragma unroll
    for (int nf = 0; nf < 2; ++nf) {
#pragma unroll
      for (int j = 0; j < 8; ++j) {
        const int grow = mbase + wm * 64 + mf * 16 + lhalf * 8 + j;  // bs
        const int gcol = nbase + wn * 32 + nf * 16 + lrow;           // he
        const float v = acc[mf][nf][j];
        const size_t idx = (size_t)grow * NHE + gcol;
        if (sel == 0) {
          qb[idx] = (bf16)(v * 0.08838834764831845f);  // fold 1/sqrt(d_head)
        } else if (sel == 1) {
          kb[idx] = (bf16)v;
          kout[idx] = v;
        } else {
          vout[idx] = v;
          const int bi = grow >> 11, si = grow & (NS - 1);
          const int hi = gcol >> 7,  ei = gcol & (NE - 1);
          vT[(((size_t)(bi * NH + hi) * NE + ei) << 11) + si] = (bf16)v;
        }
      }
    }
  }
}

// ---------------------------------------------------------------------------
// Flash attention: block = (64 q-rows, one (b,h)); 4 waves x 16 q-rows.
// KV tile = 32 columns; online softmax; causal loop bound halves the work.
// ---------------------------------------------------------------------------
__global__ __launch_bounds__(128) void flash_attn_kernel(
    const bf16* __restrict__ qb, const bf16* __restrict__ kb,
    const bf16* __restrict__ vT, bf16* __restrict__ attnb) {
  __shared__ bf16 pshm[4][16 * LSTR];
  const int h = blockIdx.y, b = blockIdx.z;
  const int lane = threadIdx.x & 31, wid = threadIdx.x >> 5;
  const int lrow = lane & 15, lhalf = lane >> 4;
  const int qbase = blockIdx.x * 64 + wid * 16;

  // Q fragments (A-operand layout), K-dim = e (128) in 4 chunks of 32
  v16bf qf[4];
#pragma unroll
  for (int c = 0; c < 4; ++c) {
    const bf16* p =
        qb + (size_t)(b * NS + qbase + lrow) * NHE + h * NE + c * 32 + lhalf * 16;
    qf[c] = cat16(*(const v8bf*)p, *(const v8bf*)(p + 8));
  }

  v8f oacc[8] = {};   // O tile 16 x 128 (8 e-chunks)
  float m[8], l[8];
#pragma unroll
  for (int j = 0; j < 8; ++j) { m[j] = -3.0e38f; l[j] = 0.0f; }

  const int ntiles = blockIdx.x * 2 + 2;  // causal: kv tiles covering rows <= qmax
  bf16* pw = &pshm[wid][0];

  for (int t = 0; t < ntiles; ++t) {
    const int kvb = t * 32;
    // ---- S = Q * K^T (16 x 32 tile, f32) ----
    v8f sacc[2] = {};
#pragma unroll
    for (int nc = 0; nc < 2; ++nc) {
      const int kvrow = kvb + nc * 16 + lrow;  // B-operand N = kv row
      const bf16* kp = kb + (size_t)(b * NS + kvrow) * NHE + h * NE;
#pragma unroll
      for (int c = 0; c < 4; ++c) {
        const bf16* pk = kp + c * 32 + lhalf * 16;  // B layout: contiguous K half
        v16bf kf = cat16(*(const v8bf*)pk, *(const v8bf*)(pk + 8));
        sacc[nc] = wmma_bf16(qf[c], kf, sacc[nc]);
      }
    }
    // ---- causal mask + online softmax ----
    const int col0 = kvb + lrow;
    const int col1 = kvb + 16 + lrow;
    const int rbase = qbase + lhalf * 8;
#pragma unroll
    for (int j = 0; j < 8; ++j) {
      const int rowg = rbase + j;
      const float s0 = (col0 <= rowg) ? sacc[0][j] : -3.0e38f;
      const float s1 = (col1 <= rowg) ? sacc[1][j] : -3.0e38f;
      float mx = fmaxf(s0, s1);
      mx = fmaxf(mx, __shfl_xor(mx, 1, 16));
      mx = fmaxf(mx, __shfl_xor(mx, 2, 16));
      mx = fmaxf(mx, __shfl_xor(mx, 4, 16));
      mx = fmaxf(mx, __shfl_xor(mx, 8, 16));
      const float mn = fmaxf(m[j], mx);
      const float p0 = (s0 > -1.0e38f) ? __expf(s0 - mn) : 0.0f;
      const float p1 = (s1 > -1.0e38f) ? __expf(s1 - mn) : 0.0f;
      const float rs = __expf(m[j] - mn);
      float sum = p0 + p1;
      sum += __shfl_xor(sum, 1, 16);
      sum += __shfl_xor(sum, 2, 16);
      sum += __shfl_xor(sum, 4, 16);
      sum += __shfl_xor(sum, 8, 16);
      l[j] = l[j] * rs + sum;
      m[j] = mn;
#pragma unroll
      for (int ec = 0; ec < 8; ++ec) oacc[ec][j] *= rs;
      // stash P (C-layout position) into per-wave LDS tile as bf16
      pw[(lhalf * 8 + j) * LSTR + lrow] = (bf16)p0;
      pw[(lhalf * 8 + j) * LSTR + 16 + lrow] = (bf16)p1;
    }
    // ---- reload P in A-operand layout, O += P * V ----
    const bf16* pp = pw + lrow * LSTR + lhalf * 16;
    v16bf pf = cat16(*(const v8bf*)pp, *(const v8bf*)(pp + 8));
#pragma unroll
    for (int ec = 0; ec < 8; ++ec) {
      const int e = ec * 16 + lrow;  // B-operand N = e
      const bf16* vp = vT + ((size_t)(b * NH + h) * NE + e) * NS + kvb + lhalf * 16;
      v16bf vf = cat16(*(const v8bf*)vp, *(const v8bf*)(vp + 8));
      oacc[ec] = wmma_bf16(pf, vf, oacc[ec]);
    }
  }
  // ---- finalize: O / l, store bf16 attn_out [b,s,h,e] ----
#pragma unroll
  for (int j = 0; j < 8; ++j) {
    const float inv = 1.0f / l[j];
    const int qg = qbase + lhalf * 8 + j;
    bf16* op = attnb + (size_t)(b * NS + qg) * NHE + h * NE;
#pragma unroll
    for (int ec = 0; ec < 8; ++ec) op[ec * 16 + lrow] = (bf16)(oacc[ec][j] * inv);
  }
}

// ---------------------------------------------------------------------------
// Output projection: out[bs][d] = attn[bs][he] * WoT[d][he]^T  (f32 store)
// ---------------------------------------------------------------------------
__global__ __launch_bounds__(256) void oproj_gemm_kernel(
    const bf16* __restrict__ A, const bf16* __restrict__ Bt,
    float* __restrict__ out) {
  const int mbase = blockIdx.x * 128, nbase = blockIdx.y * 128;
  v8f acc[4][2] = {};
  gemm_core_2048(A, Bt, mbase, nbase, acc);
  const int lane = threadIdx.x & 31, wid = threadIdx.x >> 5;
  const int lrow = lane & 15, lhalf = lane >> 4;
  const int wm = wid & 1, wn = wid >> 1;
#pragma unroll
  for (int mf = 0; mf < 4; ++mf)
#pragma unroll
    for (int nf = 0; nf < 2; ++nf)
#pragma unroll
      for (int j = 0; j < 8; ++j) {
        const int grow = mbase + wm * 64 + mf * 16 + lhalf * 8 + j;
        const int gcol = nbase + wn * 32 + nf * 16 + lrow;
        out[(size_t)grow * ND + gcol] = acc[mf][nf][j];
      }
}

// ---------------------------------------------------------------------------
extern "C" void kernel_launch(void* const* d_in, const int* in_sizes, int n_in,
                              void* d_out, int out_size, void* d_ws, size_t ws_size,
                              hipStream_t stream) {
  (void)in_sizes; (void)n_in; (void)out_size; (void)ws_size;
  const float* resid = (const float*)d_in[0];
  const float* WQ = (const float*)d_in[1];
  const float* WK = (const float*)d_in[2];
  const float* WV = (const float*)d_in[3];
  const float* WO = (const float*)d_in[4];
  float* out = (float*)d_out;
  const size_t BSD = (size_t)NB * NS * ND;   // 16,777,216
  const size_t WSZ = (size_t)NH * NE * ND;   //  4,194,304
  float* kout = out + BSD;
  float* vout = out + 2 * BSD;

  // workspace layout (bf16), ~201 MB total
  bf16* p = (bf16*)d_ws;
  bf16* residb = p; p += BSD;
  bf16* wqb    = p; p += WSZ;
  bf16* wkb    = p; p += WSZ;
  bf16* wvb    = p; p += WSZ;
  bf16* woTb   = p; p += WSZ;
  bf16* qbuf   = p; p += BSD;
  bf16* kbuf   = p; p += BSD;
  bf16* vTbuf  = p; p += BSD;
  bf16* attnb  = p; p += BSD;

  cvt_bf16_kernel<<<(int)(BSD / 4 / 256), 256, 0, stream>>>(resid, residb, (int)(BSD / 4));
  cvt_bf16_kernel<<<(int)(WSZ / 4 / 256), 256, 0, stream>>>(WQ, wqb, (int)(WSZ / 4));
  cvt_bf16_kernel<<<(int)(WSZ / 4 / 256), 256, 0, stream>>>(WK, wkb, (int)(WSZ / 4));
  cvt_bf16_kernel<<<(int)(WSZ / 4 / 256), 256, 0, stream>>>(WV, wvb, (int)(WSZ / 4));
  cvt_t_kernel<<<dim3(ND / 32, NHE / 32), dim3(32, 8), 0, stream>>>(WO, woTb);

  qkv_gemm_kernel<<<dim3(NB * NS / 128, NHE / 128, 3), 256, 0, stream>>>(
      residb, wqb, wkb, wvb, qbuf, kbuf, vTbuf, kout, vout);

  flash_attn_kernel<<<dim3(NS / 64, NH, NB), 128, 0, stream>>>(qbuf, kbuf, vTbuf, attnb);

  oproj_gemm_kernel<<<dim3(NB * NS / 128, ND / 128), 256, 0, stream>>>(attnb, woTb, out);
}
